// ESNCell_26431228740045
// MI455X (gfx1250) — matrix operand order repaired and checked by reading, not compiled, not run on hardware
//
#include <hip/hip_runtime.h>
#include <hip/hip_bf16.h>

#define T_STEPS   4096
#define IN_DIM    512
#define HID       2048
#define LEAK      0.3f

// ---- WMMA types (CDNA5 / gfx1250, wave32) ----
typedef __bf16 v16bf __attribute__((ext_vector_type(16)));
typedef float  v8f   __attribute__((ext_vector_type(8)));

// =====================================================================
// Phase 1: P[t][n] = sum_k U[t][k] * Win[n][k]   (GEMM M=4096,N=2048,K=512)
// One wave per 32x32 output block = 2x2 register-blocked 16x16 WMMA tiles,
// v_wmma_f32_16x16x32_bf16 over K (4 WMMAs per K-step, A/B fragments each
// reused twice).
// A frag (16x32 bf16): lane L<16 -> row L, K {kb..kb+7, kb+16..kb+23}, kb=(L>>4)*8
// B frag (32x16 bf16): lane L    -> col L&15, K {kb..kb+7, kb+16..kb+23}, kb=(L>>4)*8
//   (B[k][n] = Win[n][k], so B columns are contiguous rows of Win)
// C/D (16x16 f32): lane L -> col L&15, VGPR r -> row r + (L>=16 ? 8 : 0)
// =====================================================================
__global__ __launch_bounds__(256) void esn_proj_wmma(
    const float* __restrict__ U, const float* __restrict__ Win,
    float* __restrict__ P)
{
    const int wave = threadIdx.x >> 5;
    const int lane = threadIdx.x & 31;
    const int job  = blockIdx.x * 8 + wave;     // 8192 jobs: 128 mpairs x 64 npairs
    const int mpair = job >> 6;                 // / (HID/32)
    const int npair = job & 63;

    const int arow = lane & 15;
    const int akb  = (lane >> 4) << 3;          // 0 or 8
    const int bcol = lane & 15;
    const int bkb  = (lane >> 4) << 3;          // 0 or 8

    const float* __restrict__ Ua0 = U   + (size_t)(mpair * 32 + arow)      * IN_DIM;
    const float* __restrict__ Ua1 = U   + (size_t)(mpair * 32 + 16 + arow) * IN_DIM;
    const float* __restrict__ Wb0 = Win + (size_t)(npair * 32 + bcol)      * IN_DIM;
    const float* __restrict__ Wb1 = Win + (size_t)(npair * 32 + 16 + bcol) * IN_DIM;

    v8f acc00 = {}, acc01 = {}, acc10 = {}, acc11 = {};

    for (int k0 = 0; k0 < IN_DIM; k0 += 32) {
        v16bf a0, a1, b0, b1;
#pragma unroll
        for (int j = 0; j < 8; ++j) {
            a0[j]     = (__bf16)Ua0[k0 + akb + j];
            a0[j + 8] = (__bf16)Ua0[k0 + akb + 16 + j];
            a1[j]     = (__bf16)Ua1[k0 + akb + j];
            a1[j + 8] = (__bf16)Ua1[k0 + akb + 16 + j];
        }
#pragma unroll
        for (int j = 0; j < 8; ++j) {
            b0[j]     = (__bf16)Wb0[k0 + bkb + j];
            b0[j + 8] = (__bf16)Wb0[k0 + bkb + 16 + j];
            b1[j]     = (__bf16)Wb1[k0 + bkb + j];
            b1[j + 8] = (__bf16)Wb1[k0 + bkb + 16 + j];
        }
        acc00 = __builtin_amdgcn_wmma_f32_16x16x32_bf16(false, a0, false, b0, (short)0, acc00, false, false);
        acc01 = __builtin_amdgcn_wmma_f32_16x16x32_bf16(false, a0, false, b1, (short)0, acc01, false, false);
        acc10 = __builtin_amdgcn_wmma_f32_16x16x32_bf16(false, a1, false, b0, (short)0, acc10, false, false);
        acc11 = __builtin_amdgcn_wmma_f32_16x16x32_bf16(false, a1, false, b1, (short)0, acc11, false, false);
    }

    const int ncol  = lane & 15;
    const int mbase = ((lane >> 4) << 3);       // 0 or 8 within a 16-row tile
    float* __restrict__ P00 = P + (size_t)(mpair * 32 + mbase)      * HID + npair * 32 + ncol;
    float* __restrict__ P01 = P00 + 16;
    float* __restrict__ P10 = P + (size_t)(mpair * 32 + 16 + mbase) * HID + npair * 32 + ncol;
    float* __restrict__ P11 = P10 + 16;
#pragma unroll
    for (int r = 0; r < 8; ++r) {
        P00[(size_t)r * HID] = acc00[r];
        P01[(size_t)r * HID] = acc01[r];
        P10[(size_t)r * HID] = acc10[r];
        P11[(size_t)r * HID] = acc11[r];
    }
}

// =====================================================================
// Phase 2: persistent cooperative recurrence.
// 128 workgroups x 256 threads; WG g owns 16 rows of W, cached in 128 KB
// of LDS once (full 16 MB W is LDS-resident device-wide). Per step:
// stage h_{t-1} (8 KB) into LDS, 16 conflict-free dot products
// (lane-interleaved k), shfl_xor reduction within 16-lane halves (wave32),
// tanh+leak, write slice of H[t], grid barrier (monotonic atomic count).
// d_out (H) itself carries the h chain between steps.
// =====================================================================
#define GWG  128
#define RPG  (HID / GWG)   // 16 rows per workgroup

__global__ __launch_bounds__(256) void esn_recur(
    const float* __restrict__ W, const float* __restrict__ bias,
    const float* __restrict__ P, float* __restrict__ H,
    unsigned* bar_count, unsigned* bar_gen)
{
    extern __shared__ float smem[];
    float* Wl = smem;              // RPG * HID floats (128 KB)
    float* hl = smem + RPG * HID;  // HID floats (8 KB)

    const int tid = threadIdx.x;
    const int wg  = blockIdx.x;
    const int r   = tid >> 4;      // row within slice: 0..15
    const int c   = tid & 15;      // k-phase within row group

    // Stage this WG's W slice into LDS once.
    const float* __restrict__ Wsrc = W + (size_t)wg * RPG * HID;
    for (int i = tid; i < RPG * HID; i += 256) Wl[i] = Wsrc[i];
    __syncthreads();

    for (int t = 0; t < T_STEPS; ++t) {
        // Stage h_{t-1} (zeros for t==0) into LDS.
        if (t == 0) {
            for (int i = tid; i < HID; i += 256) hl[i] = 0.0f;
        } else {
            const float* __restrict__ hp = H + (size_t)(t - 1) * HID;
            for (int i = tid; i < HID; i += 256) hl[i] = hp[i];
        }
        __syncthreads();

        // Partial dot: row (wg*RPG + r), k = c + 16*j  (consecutive lanes ->
        // consecutive LDS banks for W; h reads are 2-lane broadcasts).
        float sum = 0.0f;
        const float* wrow = Wl + r * HID;
#pragma unroll 8
        for (int j = 0; j < HID / 16; ++j) {
            const int k = c + (j << 4);
            sum = fmaf(wrow[k], hl[k], sum);
        }
        // Reduce 16 partials within each 16-lane half (wave32).
        sum += __shfl_xor(sum, 8, 16);
        sum += __shfl_xor(sum, 4, 16);
        sum += __shfl_xor(sum, 2, 16);
        sum += __shfl_xor(sum, 1, 16);

        if (c == 0) {
            const int n    = wg * RPG + r;
            const float pre  = sum + P[(size_t)t * HID + n] + bias[n];
            const float hnew = (1.0f - LEAK) * hl[n] + LEAK * tanhf(pre);
            H[(size_t)t * HID + n] = hnew;
        }
        __syncthreads();  // all lanes done with hl before next stage / leave

        // Grid-wide barrier: monotonic arrival count, gen = #completed steps.
        if (tid == 0) {
            __threadfence();
            const unsigned target = (unsigned)(t + 1) * GWG;
            const unsigned arrived = atomicAdd(bar_count, 1u) + 1u;
            if (arrived == target) {
                __threadfence();
                atomicAdd(bar_gen, 1u);
            } else {
                while (__atomic_load_n(bar_gen, __ATOMIC_ACQUIRE) < (unsigned)(t + 1)) { }
            }
        }
        __syncthreads();
    }
}

extern "C" void kernel_launch(void* const* d_in, const int* in_sizes, int n_in,
                              void* d_out, int out_size, void* d_ws, size_t ws_size,
                              hipStream_t stream) {
    const float* U    = (const float*)d_in[0];   // (4096, 512)
    const float* Win  = (const float*)d_in[1];   // (2048, 512)
    const float* W    = (const float*)d_in[2];   // (2048, 2048)
    const float* bias = (const float*)d_in[3];   // (2048,)
    float* H = (float*)d_out;                    // (4096, 2048)

    unsigned char* ws = (unsigned char*)d_ws;
    unsigned* bar_count = (unsigned*)ws;
    unsigned* bar_gen   = (unsigned*)(ws + 8);
    float*    P         = (float*)(ws + 256);    // 4096*2048 fp32 = 32 MB

    // Re-arm barrier state every call (graph-capture safe).
    hipMemsetAsync(ws, 0, 256, stream);

    // Phase 1: P = U @ Win^T via bf16 WMMA, 32x32 per wave (2x2 tiles).
    // 8192 wave-jobs / 8 waves per block = 1024 blocks.
    esn_proj_wmma<<<1024, 256, 0, stream>>>(U, Win, P);

    // Phase 2: persistent recurrence. Dynamic LDS: W slice + h vector.
    const size_t ldsBytes = (size_t)(RPG * HID + HID) * sizeof(float);
    esn_recur<<<GWG, 256, ldsBytes, stream>>>(W, bias, P, H, bar_count, bar_gen);
}
